// SimpleMultiheadAttention_78950088835257
// MI455X (gfx1250) — matrix operand order repaired and checked by reading, not compiled, not run on hardware
//
#include <hip/hip_runtime.h>
#include <hip/hip_bf16.h>

#define EMBED 1024
#define NHEAD 16
#define HDIM  64
#define BATCH 4
#define SEQ   2048
#define MROWS (BATCH*SEQ)   // 8192

typedef __attribute__((ext_vector_type(16))) __bf16         v16bf;
typedef __attribute__((ext_vector_type(8)))  float          v8f;
typedef __attribute__((ext_vector_type(8)))  unsigned short v8u;
typedef __attribute__((ext_vector_type(4)))  float          f4;

#if __has_builtin(__builtin_amdgcn_global_load_async_to_lds_b128)
#define HAVE_ASYNC_LDS 1
#else
#define HAVE_ASYNC_LDS 0
#endif

struct Frag { v8u lo, hi; };

__device__ __forceinline__ v16bf make_frag(v8u lo, v8u hi) {
    Frag f; f.lo = lo; f.hi = hi;
    return __builtin_bit_cast(v16bf, f);
}

// fp32 -> bf16 bits, round-to-nearest-even
__device__ __forceinline__ unsigned short f2bf(float x) {
    unsigned u = __float_as_uint(x);
    u += 0x7FFFu + ((u >> 16) & 1u);
    return (unsigned short)(u >> 16);
}

#if HAVE_ASYNC_LDS
// The builtin wants int4 pointers: AS(1) for global, AS(3) for LDS.
typedef int v4i __attribute__((vector_size(16)));
// generic LDS pointer -> AS(3): LDS aperture maps addr[31:0] to the LDS offset
__device__ __forceinline__ __attribute__((address_space(3))) v4i* lds4(void* p) {
    return (__attribute__((address_space(3))) v4i*)(unsigned int)(unsigned long long)p;
}
__device__ __forceinline__ __attribute__((address_space(1))) v4i* glb4(const void* p) {
    return (__attribute__((address_space(1))) v4i*)(unsigned long long)p;
}
#endif

// ---------------------------------------------------------------------------
// GEMM: out[m,n] = sum_k X[m,k] * W[n,k] + bias[n]   (nn.Linear: x @ W.T + b)
// M=8192, N=K=1024. Workgroup tile 128x128, BK=64, 8 waves of 32x64 each:
// 16 WMMAs per staged tile per wave between barriers.
// ---------------------------------------------------------------------------
template<bool IN_BF16, bool QKV_OUT>
__global__ __launch_bounds__(256)
void gemm_bias_kernel(const void* __restrict__ Xv, const float* __restrict__ W,
                      const float* __restrict__ bias, void* __restrict__ outv)
{
    __shared__ unsigned short As[128][72];   // bf16 bits, 64 cols + 8 pad (144B rows)
    __shared__ unsigned short Bs[128][72];

    const int tid  = threadIdx.x;
    const int lane = tid & 31, hl = lane >> 4, l16 = lane & 15;
    const int wave = tid >> 5;
    const int wm = wave & 3, wn = wave >> 2;         // 4 waves along M, 2 along N
    const int m0 = blockIdx.y * 128, n0 = blockIdx.x * 128;

    v8f acc[2][4] = {};

    for (int k0 = 0; k0 < EMBED; k0 += 64) {
        __syncthreads();
        // ---- stage A tile (128x64) into LDS as bf16 ----
        if (IN_BF16) {
            const unsigned short* X = (const unsigned short*)Xv;
            #pragma unroll
            for (int i = 0; i < 4; ++i) {
                int c = tid + i * 256;                    // 1024 chunks of 8
                int row = c >> 3, col = (c & 7) * 8;
                const unsigned short* src = &X[(size_t)(m0 + row) * EMBED + k0 + col];
                *(v8u*)&As[row][col] = *(const v8u*)src;
                if (k0 + 64 < EMBED) __builtin_prefetch(src + 64, 0, 3);
            }
        } else {
            const float* X = (const float*)Xv;
            #pragma unroll
            for (int i = 0; i < 8; ++i) {
                int c = tid + i * 256;                    // 2048 chunks of 4
                int row = c >> 4, col = (c & 15) * 4;
                const float* src = &X[(size_t)(m0 + row) * EMBED + k0 + col];
                f4 xv = *(const f4*)src;
                if (k0 + 64 < EMBED) __builtin_prefetch(src + 64, 0, 3);
                unsigned short* d = &As[row][col];
                d[0] = f2bf(xv.x); d[1] = f2bf(xv.y); d[2] = f2bf(xv.z); d[3] = f2bf(xv.w);
            }
        }
        // ---- stage B tile: rows of W (n0..n0+127, k0..k0+63) ----
        #pragma unroll
        for (int i = 0; i < 8; ++i) {
            int c = tid + i * 256;
            int row = c >> 4, col = (c & 15) * 4;
            const float* src = &W[(size_t)(n0 + row) * EMBED + k0 + col];
            f4 wv = *(const f4*)src;
            if (k0 + 64 < EMBED) __builtin_prefetch(src + 64, 0, 3);
            unsigned short* d = &Bs[row][col];
            d[0] = f2bf(wv.x); d[1] = f2bf(wv.y); d[2] = f2bf(wv.z); d[3] = f2bf(wv.w);
        }
        __syncthreads();

        // ---- two K=32 steps per staged tile, 8 WMMAs each ----
        #pragma unroll
        for (int ks = 0; ks < 2; ++ks) {
            v16bf a[2], b[4];
            #pragma unroll
            for (int mi = 0; mi < 2; ++mi) {
                const unsigned short* p = &As[wm * 32 + mi * 16 + l16][ks * 32 + hl * 8];
                a[mi] = make_frag(*(const v8u*)p, *(const v8u*)(p + 16));
            }
            #pragma unroll
            for (int ni = 0; ni < 4; ++ni) {
                const unsigned short* p = &Bs[wn * 64 + ni * 16 + l16][ks * 32 + hl * 8];
                b[ni] = make_frag(*(const v8u*)p, *(const v8u*)(p + 16));
            }
            #pragma unroll
            for (int mi = 0; mi < 2; ++mi)
                #pragma unroll
                for (int ni = 0; ni < 4; ++ni)
                    acc[mi][ni] = __builtin_amdgcn_wmma_f32_16x16x32_bf16(
                        false, a[mi], false, b[ni], (short)0, acc[mi][ni], false, false);
        }
    }

    // ---- epilogue: C/D layout VGPR r -> M = r + 8*hl, N = l16 ----
    #pragma unroll
    for (int mi = 0; mi < 2; ++mi) {
        #pragma unroll
        for (int ni = 0; ni < 4; ++ni) {
            #pragma unroll
            for (int r = 0; r < 8; ++r) {
                int m = m0 + wm * 32 + mi * 16 + r + hl * 8;
                int n = n0 + wn * 64 + ni * 16 + l16;
                float val = acc[mi][ni][r] + bias[n];
                if (QKV_OUT) {
                    unsigned short* out = (unsigned short*)outv;
                    int bb = m >> 11, tt = m & (SEQ - 1);
                    int hh = n >> 6,  dd = n & (HDIM - 1);
                    out[(((size_t)(bb * NHEAD + hh)) * SEQ + tt) * HDIM + dd] = f2bf(val);
                } else {
                    float* out = (float*)outv;
                    out[(size_t)m * EMBED + n] = val;
                }
            }
        }
    }
}

// ---------------------------------------------------------------------------
// Flash-attention: Q,K,V bf16 [B*H, T, 64]. One 16-row Q tile per wave,
// key blocks of 64 staged in LDS (K via async-to-LDS DMA when available,
// V transposed manually). Online softmax. Output bf16 [B, T, EMBED].
// ---------------------------------------------------------------------------
__global__ __launch_bounds__(256)
void attention_kernel(const unsigned short* __restrict__ Q,
                      const unsigned short* __restrict__ Kt,
                      const unsigned short* __restrict__ V,
                      unsigned short* __restrict__ Out)
{
    __shared__ unsigned short Klds[64][72];        // [key][d]
    __shared__ unsigned short Vlds[64][136];       // [d][key]  (transposed)
    __shared__ unsigned short Plds[8][16][136];    // per-wave P scratch

    const int tid  = threadIdx.x;
    const int lane = tid & 31, hl = lane >> 4, l16 = lane & 15;
    const int wave = tid >> 5;
    const int bh = blockIdx.y;                     // b*NHEAD + h
    const int q0 = blockIdx.x * 128 + wave * 16;
    const size_t base = (size_t)bh * SEQ * HDIM;

    // Q fragments: 16 rows x 64 (two K=32 steps)
    v16bf aq[2];
    #pragma unroll
    for (int f = 0; f < 2; ++f) {
        const unsigned short* p = &Q[base + (size_t)(q0 + l16) * HDIM + f * 32 + hl * 8];
        aq[f] = make_frag(*(const v8u*)p, *(const v8u*)(p + 16));
    }

    float Mrow[8], Lrow[8];
    v8f o[4] = {};
    #pragma unroll
    for (int r = 0; r < 8; ++r) { Mrow[r] = -1e30f; Lrow[r] = 0.f; }

    for (int kb = 0; kb < SEQ; kb += 64) {
        __syncthreads();
        // stage K block (async DMA to LDS when available), V block transposed
        #pragma unroll
        for (int i = 0; i < 2; ++i) {
            int c = tid + i * 256;                 // 512 chunks of 8
            int row = c >> 3, col = (c & 7) * 8;
#if HAVE_ASYNC_LDS
            __builtin_amdgcn_global_load_async_to_lds_b128(
                glb4(&Kt[base + (size_t)(kb + row) * HDIM + col]),
                lds4(&Klds[row][col]), 0, 0);
#else
            *(v8u*)&Klds[row][col] =
                *(const v8u*)&Kt[base + (size_t)(kb + row) * HDIM + col];
#endif
            v8u vv = *(const v8u*)&V[base + (size_t)(kb + row) * HDIM + col];
            #pragma unroll
            for (int j = 0; j < 8; ++j) Vlds[col + j][row] = vv[j];
        }
#if HAVE_ASYNC_LDS
#if __has_builtin(__builtin_amdgcn_s_wait_asynccnt)
        __builtin_amdgcn_s_wait_asynccnt(0);
#else
        asm volatile("s_wait_asynccnt 0x0" ::: "memory");
#endif
#endif
        __syncthreads();

        // S = Q * K^T  (4 n-tiles x 2 k-steps)
        v8f s[4] = {};
        #pragma unroll
        for (int j = 0; j < 4; ++j) {
            #pragma unroll
            for (int t = 0; t < 2; ++t) {
                const unsigned short* p = &Klds[j * 16 + l16][t * 32 + hl * 8];
                v16bf bk = make_frag(*(const v8u*)p, *(const v8u*)(p + 16));
                s[j] = __builtin_amdgcn_wmma_f32_16x16x32_bf16(
                    false, aq[t], false, bk, (short)0, s[j], false, false);
            }
        }

        // online softmax per row (row r+8*hl lives in this lane-half)
        #pragma unroll
        for (int r = 0; r < 8; ++r) {
            float mx = Mrow[r];
            #pragma unroll
            for (int j = 0; j < 4; ++j) {
                float v = s[j][r] * 0.125f;        // 1/sqrt(64)
                s[j][r] = v;
                #pragma unroll
                for (int m = 1; m <= 8; m <<= 1) v = fmaxf(v, __shfl_xor(v, m, 32));
                mx = fmaxf(mx, v);
            }
            float alpha = __expf(Mrow[r] - mx);
            Mrow[r] = mx;
            Lrow[r] *= alpha;
            #pragma unroll
            for (int dt = 0; dt < 4; ++dt) o[dt][r] *= alpha;
            float rsum = 0.f;
            #pragma unroll
            for (int j = 0; j < 4; ++j) {
                float pv = __expf(s[j][r] - mx);
                s[j][r] = pv;
                float ss = pv;
                #pragma unroll
                for (int m = 1; m <= 8; m <<= 1) ss += __shfl_xor(ss, m, 32);
                rsum += ss;
            }
            Lrow[r] += rsum;
            // spill P (D-layout) to per-wave LDS as bf16
            #pragma unroll
            for (int j = 0; j < 4; ++j)
                Plds[wave][r + hl * 8][j * 16 + l16] = f2bf(s[j][r]);
        }

        // reload P in A-fragment layout; O += P * V
        v16bf ap[2];
        #pragma unroll
        for (int t = 0; t < 2; ++t) {
            const unsigned short* p = &Plds[wave][l16][t * 32 + hl * 8];
            ap[t] = make_frag(*(const v8u*)p, *(const v8u*)(p + 16));
        }
        #pragma unroll
        for (int dt = 0; dt < 4; ++dt) {
            #pragma unroll
            for (int t = 0; t < 2; ++t) {
                const unsigned short* p = &Vlds[dt * 16 + l16][t * 32 + hl * 8];
                v16bf bv = make_frag(*(const v8u*)p, *(const v8u*)(p + 16));
                o[dt] = __builtin_amdgcn_wmma_f32_16x16x32_bf16(
                    false, ap[t], false, bv, (short)0, o[dt], false, false);
            }
        }
    }

    // write [b][t][h*64+d] bf16 for the final projection GEMM
    const int b = bh >> 4, h = bh & 15;
    #pragma unroll
    for (int dt = 0; dt < 4; ++dt) {
        #pragma unroll
        for (int r = 0; r < 8; ++r) {
            int tq = q0 + r + hl * 8;
            float val = o[dt][r] / Lrow[r];
            Out[((size_t)(b * SEQ + tq)) * EMBED + h * HDIM + dt * 16 + l16] = f2bf(val);
        }
    }
}

// ---------------------------------------------------------------------------
extern "C" void kernel_launch(void* const* d_in, const int* in_sizes, int n_in,
                              void* d_out, int out_size, void* d_ws, size_t ws_size,
                              hipStream_t stream)
{
    const float* q_in = (const float*)d_in[0];
    const float* k_in = (const float*)d_in[1];
    const float* v_in = (const float*)d_in[2];
    const float* Wq = (const float*)d_in[3];  const float* bq = (const float*)d_in[4];
    const float* Wk = (const float*)d_in[5];  const float* bk = (const float*)d_in[6];
    const float* Wv = (const float*)d_in[7];  const float* bv = (const float*)d_in[8];
    const float* Wo = (const float*)d_in[9];  const float* bo = (const float*)d_in[10];

    const size_t elems = (size_t)MROWS * EMBED;          // 8M elements
    unsigned short* qws = (unsigned short*)d_ws;          // 16 MB each
    unsigned short* kws = qws + elems;
    unsigned short* vws = kws + elems;
    unsigned short* aws = vws + elems;                    // attn out bf16 [B,T,EMBED]

    dim3 blk(256);
    dim3 gemm_grid(EMBED / 128, MROWS / 128);             // (8, 64)

    gemm_bias_kernel<false, true><<<gemm_grid, blk, 0, stream>>>(q_in, Wq, bq, qws);
    gemm_bias_kernel<false, true><<<gemm_grid, blk, 0, stream>>>(k_in, Wk, bk, kws);
    gemm_bias_kernel<false, true><<<gemm_grid, blk, 0, stream>>>(v_in, Wv, bv, vws);

    dim3 attn_grid(SEQ / 128, BATCH * NHEAD);             // (16, 64)
    attention_kernel<<<attn_grid, blk, 0, stream>>>(qws, kws, vws, aws);

    gemm_bias_kernel<true, false><<<gemm_grid, blk, 0, stream>>>(aws, Wo, bo, d_out);
}
